// Lost_14121852469570
// MI455X (gfx1250) — matrix operand order repaired and checked by reading, not compiled, not run on hardware
//
#include <hip/hip_runtime.h>

typedef __attribute__((ext_vector_type(16))) __bf16 v16bf;
typedef __attribute__((ext_vector_type(8)))  __bf16 v8bf;
typedef __attribute__((ext_vector_type(8)))  float  v8f;

#define N_PATCH 4096
#define DIM     768
#define KTILE   32
#define TOPK    100

// float -> bf16 round-to-nearest-even (raw u16 storage)
__device__ __forceinline__ unsigned short f2bf(float f) {
  unsigned int u = __float_as_uint(f);
  u += 0x7fffu + ((u >> 16) & 1u);
  return (unsigned short)(u >> 16);
}

// ---------------- patchify: img[3,1024,1024] -> patches[4096,768] bf16 ----------------
__global__ void patchify_kernel(const float* __restrict__ img,
                                unsigned short* __restrict__ patches) {
  int idx = blockIdx.x * blockDim.x + threadIdx.x;      // 4096*768 threads
  int p = idx / DIM, j = idx - p * DIM;
  int c = j % 3, pix = j / 3;
  int py = pix >> 4, px = pix & 15;
  int hd = p >> 6,  wd = p & 63;
  int y = hd * 16 + py, x = wd * 16 + px;
  patches[idx] = f2bf(img[((size_t)c * 1024 + y) * 1024 + x]);
}

// ---------------- W[768,768] (K,N) -> Wt[768,768] (N,K) bf16 ----------------
__global__ void wtrans_kernel(const float* __restrict__ W,
                              unsigned short* __restrict__ Wt) {
  int idx = blockIdx.x * blockDim.x + threadIdx.x;      // 768*768 threads
  int n = idx / DIM, k = idx - n * DIM;
  Wt[(size_t)n * DIM + k] = f2bf(W[(size_t)k * DIM + n]);
}

// ---------------- bf16 WMMA GEMM: C[M,N] = A[M,K] * B[N,K]^T ----------------
// Per wave: 32x64 C tile = 2x4 v_wmma_f32_16x16x32_bf16 per K-step, software-
// pipelined with double-buffered fragments. Addressing = uniform SGPR base +
// one 32-bit per-lane offset VGPR per row (k-advance folds into the SGPR base),
// so no 64-bit per-lane pointer state -> no spills. Block = 8 waves = 128x128.

struct Frags {
  v16bf a0, a1;          // two A 16x32 fragments (M rows mbase / mbase+16)
  v16bf b0, b1, b2, b3;  // four B 32x16 fragments (N cols nbase+0/16/32/48)
};

__device__ __forceinline__ v16bf cat8(v8bf lo, v8bf hi) {
  return __builtin_shufflevector(lo, hi, 0, 1, 2, 3, 4, 5, 6, 7,
                                         8, 9, 10, 11, 12, 13, 14, 15);
}

__device__ __forceinline__ void load_frags(Frags& f,
                                           const __bf16* __restrict__ A,
                                           const __bf16* __restrict__ B,
                                           unsigned ao0, unsigned ao1,
                                           unsigned bo0, unsigned bo1,
                                           unsigned bo2, unsigned bo3,
                                           unsigned k) {
  // A 16x32 layout: lane half h holds K = h*16..h*16+15 (ao includes row*K + h*16)
  f.a0 = *(const v16bf*)(A + ao0 + k);
  f.a1 = *(const v16bf*)(A + ao1 + k);
  // B 32x16 layout: lane half h holds K = h*16..h*16+15 (bo includes row*K + h*16)
  f.b0 = *(const v16bf*)(B + bo0 + k);
  f.b1 = *(const v16bf*)(B + bo1 + k);
  f.b2 = *(const v16bf*)(B + bo2 + k);
  f.b3 = *(const v16bf*)(B + bo3 + k);
}

__device__ __forceinline__ void mma_step(v8f acc[2][4], const Frags& f) {
  acc[0][0] = __builtin_amdgcn_wmma_f32_16x16x32_bf16(false, f.a0, false, f.b0, (short)0, acc[0][0], false, false);
  acc[1][0] = __builtin_amdgcn_wmma_f32_16x16x32_bf16(false, f.a1, false, f.b0, (short)0, acc[1][0], false, false);
  acc[0][1] = __builtin_amdgcn_wmma_f32_16x16x32_bf16(false, f.a0, false, f.b1, (short)0, acc[0][1], false, false);
  acc[1][1] = __builtin_amdgcn_wmma_f32_16x16x32_bf16(false, f.a1, false, f.b1, (short)0, acc[1][1], false, false);
  acc[0][2] = __builtin_amdgcn_wmma_f32_16x16x32_bf16(false, f.a0, false, f.b2, (short)0, acc[0][2], false, false);
  acc[1][2] = __builtin_amdgcn_wmma_f32_16x16x32_bf16(false, f.a1, false, f.b2, (short)0, acc[1][2], false, false);
  acc[0][3] = __builtin_amdgcn_wmma_f32_16x16x32_bf16(false, f.a0, false, f.b3, (short)0, acc[0][3], false, false);
  acc[1][3] = __builtin_amdgcn_wmma_f32_16x16x32_bf16(false, f.a1, false, f.b3, (short)0, acc[1][3], false, false);
}

template <bool STORE_BF16>
__global__ void __launch_bounds__(256)
gemm_bf16_wmma(const unsigned short* __restrict__ Aq,
               const unsigned short* __restrict__ Bq,
               void* __restrict__ Cv,
               int K, int ldc) {
  const __bf16* A = (const __bf16*)Aq;
  const __bf16* B = (const __bf16*)Bq;

  int tid  = threadIdx.x;
  int lane = tid & 31;
  int wave = tid >> 5;                 // 0..7
  int wm = wave & 3, wn = wave >> 2;   // 4 (M) x 2 (N) wave grid
  int mbase = blockIdx.x * 128 + wm * 32;
  int nbase = blockIdx.y * 128 + wn * 64;

  int hb  = (lane >> 4) & 1;           // lane half (ISA 16-bit A/B layout)
  int l15 = lane & 15;

  v8f acc[2][4];
#pragma unroll
  for (int i = 0; i < 2; i++)
#pragma unroll
    for (int j = 0; j < 4; j++) acc[i][j] = v8f{};

  // Per-lane 32-bit element offsets (one VGPR each); k advances via SGPR base.
  unsigned ao0 = (unsigned)(mbase + l15) * K      + hb * 16;
  unsigned ao1 = (unsigned)(mbase + 16 + l15) * K + hb * 16;
  unsigned bo0 = (unsigned)(nbase + 0  + l15) * K + hb * 16;
  unsigned bo1 = (unsigned)(nbase + 16 + l15) * K + hb * 16;
  unsigned bo2 = (unsigned)(nbase + 32 + l15) * K + hb * 16;
  unsigned bo3 = (unsigned)(nbase + 48 + l15) * K + hb * 16;

  // Software pipeline over S = K/KTILE steps (S even, >= 2 for both GEMMs).
  Frags f0, f1;
  load_frags(f0, A, B, ao0, ao1, bo0, bo1, bo2, bo3, 0);
  unsigned k = KTILE;
  for (; k + 2 * KTILE < (unsigned)K; k += 2 * KTILE) {
    load_frags(f1, A, B, ao0, ao1, bo0, bo1, bo2, bo3, k);
    mma_step(acc, f0);
    load_frags(f0, A, B, ao0, ao1, bo0, bo1, bo2, bo3, k + KTILE);
    mma_step(acc, f1);
  }
  load_frags(f1, A, B, ao0, ao1, bo0, bo1, bo2, bo3, (unsigned)K - KTILE);
  mma_step(acc, f0);
  mma_step(acc, f1);

  // C/D layout: VGPR r -> M = r + 8*half, N = col tile base + l15
  int r0 = hb ? 8 : 0;
#pragma unroll
  for (int i = 0; i < 2; i++) {
    int mrow = mbase + 16 * i + r0;
#pragma unroll
    for (int j = 0; j < 4; j++) {
      int ncol = nbase + 16 * j + l15;
      if (STORE_BF16) {
        unsigned short* C = (unsigned short*)Cv;
#pragma unroll
        for (int r = 0; r < 8; r++)
          C[(size_t)(mrow + r) * ldc + ncol] = f2bf(acc[i][j][r]);
      } else {
        float* C = (float*)Cv;
#pragma unroll
        for (int r = 0; r < 8; r++)
          C[(size_t)(mrow + r) * ldc + ncol] = acc[i][j][r];
      }
    }
  }
}

// ---------------- degree column sums: colsum[j] = #{i : sim[i][j] >= 0} ----------------
__global__ void colsum_kernel(const float* __restrict__ sim, int* __restrict__ colsum) {
  int j = blockIdx.x * blockDim.x + threadIdx.x;
  int c = 0;
  for (int i = 0; i < N_PATCH; i++) c += (sim[(size_t)i * N_PATCH + j] >= 0.0f) ? 1 : 0;
  colsum[j] = c;
}

// ---------------- seed = argmin(colsum), first-min tie rule ----------------
__global__ void argmin_kernel(const int* __restrict__ colsum, int* __restrict__ seed) {
  __shared__ int sval[1024];
  __shared__ int sidx[1024];
  int t = threadIdx.x;
  int bv = 0x7fffffff, bi = 0;
  for (int j = t; j < N_PATCH; j += 1024) {
    int v = colsum[j];
    if (v < bv || (v == bv && j < bi)) { bv = v; bi = j; }
  }
  sval[t] = bv; sidx[t] = bi;
  __syncthreads();
  for (int s = 512; s > 0; s >>= 1) {
    if (t < s) {
      if (sval[t + s] < sval[t] || (sval[t + s] == sval[t] && sidx[t + s] < sidx[t])) {
        sval[t] = sval[t + s]; sidx[t] = sidx[t + s];
      }
    }
    __syncthreads();
  }
  if (t == 0) *seed = sidx[0];
}

// ---------------- initial mask: (sim[seed] >= 0) AND rank-in-top-100 ----------------
__global__ void init_mask_kernel(const float* __restrict__ sim,
                                 const int* __restrict__ seedp,
                                 float* __restrict__ carry) {
  __shared__ float srow[N_PATCH];
  int t = threadIdx.x;
  int seed = *seedp;
  const float* row = sim + (size_t)seed * N_PATCH;
  for (int j = t; j < N_PATCH; j += blockDim.x) srow[j] = row[j];
  __syncthreads();
  int j = blockIdx.x * blockDim.x + t;
  float sj = srow[j];
  int rank = 0;                       // stable descending rank (argsort(-x))
  for (int l = 0; l < N_PATCH; l++) {
    float sl = srow[l];
    rank += ((sl > sj) || (sl == sj && l < j)) ? 1 : 0;
  }
  bool keep = rank < TOPK;
  bool on   = (sj >= 0.0f) || (j == seed);
  carry[j] = (keep && on) ? 1.0f : 0.0f;
}

// ---------------- sequential region-growing scan (single persistent block) ----------------
__global__ void scan_kernel(const float* __restrict__ sim,
                            const float* __restrict__ carry_init,
                            float* __restrict__ out) {
  __shared__ float carry[N_PATCH];
  __shared__ float wsum[32];
  int t = threadIdx.x;                // 1024 threads = 32 waves (wave32)
  int lane = t & 31, wave = t >> 5;
  for (int j = t; j < N_PATCH; j += 1024) carry[j] = carry_init[j];
  __syncthreads();

  for (int i = 0; i < N_PATCH; i++) {
    const float* row = sim + (size_t)i * N_PATCH;
    float p = 0.0f;
    for (int j = t; j < N_PATCH; j += 1024) p += row[j] * carry[j];
    for (int o = 16; o > 0; o >>= 1) p += __shfl_down(p, o, 32);
    if (lane == 0) wsum[wave] = p;
    __syncthreads();
    if (t == 0) {
      float s = 0.0f;
      for (int w = 0; w < 32; w++) s += wsum[w];
      float cur = carry[i];
      carry[i] = (cur == 0.0f) ? 0.0f : (s > 0.0f ? 1.0f : 0.0f);
    }
    __syncthreads();
  }
  for (int j = t; j < N_PATCH; j += 1024) out[j] = carry[j];
}

// ---------------- host orchestration ----------------
extern "C" void kernel_launch(void* const* d_in, const int* in_sizes, int n_in,
                              void* d_out, int out_size, void* d_ws, size_t ws_size,
                              hipStream_t stream) {
  (void)in_sizes; (void)n_in; (void)out_size; (void)ws_size;
  const float* img = (const float*)d_in[0];   // [1,3,1024,1024]
  const float* W   = (const float*)d_in[1];   // [768,768]
  float* out = (float*)d_out;                 // [64*64] fp32

  char* p = (char*)d_ws;
  auto alloc = [&](size_t bytes) -> char* {
    char* r = p;
    p += (bytes + 255) & ~(size_t)255;
    return r;
  };
  unsigned short* patches = (unsigned short*)alloc((size_t)N_PATCH * DIM * 2);
  unsigned short* Wt      = (unsigned short*)alloc((size_t)DIM * DIM * 2);
  unsigned short* feats   = (unsigned short*)alloc((size_t)N_PATCH * DIM * 2);
  float*          sim     = (float*)alloc((size_t)N_PATCH * N_PATCH * 4);
  int*            colsum  = (int*)alloc((size_t)N_PATCH * 4);
  int*            seed    = (int*)alloc(256);
  float*          carry   = (float*)alloc((size_t)N_PATCH * 4);

  patchify_kernel<<<(N_PATCH * DIM) / 256, 256, 0, stream>>>(img, patches);
  wtrans_kernel<<<(DIM * DIM) / 256, 256, 0, stream>>>(W, Wt);

  // GEMM1: feats[4096,768] (bf16 out) = patches[4096,768] @ Wt[768,768]^T
  gemm_bf16_wmma<true><<<dim3(N_PATCH / 128, DIM / 128), 256, 0, stream>>>(
      patches, Wt, feats, DIM, DIM);
  // GEMM2: sim[4096,4096] (f32 out) = feats @ feats^T
  gemm_bf16_wmma<false><<<dim3(N_PATCH / 128, N_PATCH / 128), 256, 0, stream>>>(
      feats, feats, sim, DIM, N_PATCH);

  colsum_kernel<<<N_PATCH / 256, 256, 0, stream>>>(sim, colsum);
  argmin_kernel<<<1, 1024, 0, stream>>>(colsum, seed);
  init_mask_kernel<<<N_PATCH / 256, 256, 0, stream>>>(sim, seed, carry);
  scan_kernel<<<1, 1024, 0, stream>>>(sim, carry, out);
}
